// MultiQueryAttention_83717502534081
// MI455X (gfx1250) — matrix operand order repaired and checked by reading, not compiled, not run on hardware
//
#include <hip/hip_runtime.h>

// ---------------------------------------------------------------------------
// MQA for gfx1250 (MI455X), wave32 + v_wmma_f32_16x16x32_bf16.
//
// Roofline: ~142 GFLOP total, but the reference *returns* attn
// [2,16,2048,2048] fp32 = 537MB -> memory-bound on attn traffic.
// Design: write attn exactly once (nontemporal), keep the normalized
// probabilities as a bf16 strip in LDS (320KB/WGP) and feed attn@V WMMAs
// directly from LDS. All bf16 conversions happen once in prep kernels;
// weights are pre-transposed so every WMMA fragment load is a contiguous
// 16B global_load_b128 / ds_load_b128.
// ---------------------------------------------------------------------------

typedef __attribute__((ext_vector_type(16))) __bf16 v16bf;
typedef __attribute__((ext_vector_type(8)))  __bf16 v8bf;
typedef __attribute__((ext_vector_type(4)))  __bf16 v4bf;
typedef __attribute__((ext_vector_type(8)))  float  v8f;
typedef __attribute__((ext_vector_type(4)))  float  v4f;

__device__ __forceinline__ __bf16 f2bf(float f) {
  unsigned u = __builtin_bit_cast(unsigned, f);
  u += 0x7FFFu + ((u >> 16) & 1u);               // round to nearest even
  unsigned short h = (unsigned short)(u >> 16);
  return __builtin_bit_cast(__bf16, h);
}

__device__ __forceinline__ v16bf cat8(v8bf lo, v8bf hi) {
  return __builtin_shufflevector(lo, hi, 0, 1, 2, 3, 4, 5, 6, 7,
                                 8, 9, 10, 11, 12, 13, 14, 15);
}
// Fragment = two contiguous 16B halves, `off` bf16 elements apart.
__device__ __forceinline__ v16bf ldfrag(const __bf16* p, int off) {
  return cat8(*(const v8bf*)p, *(const v8bf*)(p + off));
}

// -------------------------- prep: fp32 -> bf16 -----------------------------
__global__ __launch_bounds__(256) void cvt_bf16(const float* __restrict__ s,
                                                __bf16* __restrict__ d, long n) {
  const long i = ((long)blockIdx.x * 256 + threadIdx.x) * 4;
  if (i >= n) return;
  const v4f v = __builtin_nontemporal_load((const v4f*)(s + i));
  v4bf o;
  o[0] = f2bf(v.x); o[1] = f2bf(v.y); o[2] = f2bf(v.z); o[3] = f2bf(v.w);
  *(v4bf*)(d + i) = o;
}

// fp32 src[R][C] -> bf16 dst[C][R] (32x32 LDS tile transpose; R,C % 32 == 0)
__global__ __launch_bounds__(256) void cvt_tr_bf16(const float* __restrict__ s,
                                                   __bf16* __restrict__ d,
                                                   int R, int C) {
  __shared__ float t[32][33];
  const int c0 = blockIdx.x * 32, r0 = blockIdx.y * 32;
  const int tx = threadIdx.x & 31, ty = threadIdx.x >> 5;
  for (int i = ty; i < 32; i += 8)
    t[i][tx] = s[(long)(r0 + i) * C + c0 + tx];
  __syncthreads();
  for (int i = ty; i < 32; i += 8)
    d[(long)(c0 + i) * R + r0 + tx] = f2bf(t[tx][i]);
}

// ------------------------------- GEMM --------------------------------------
// C[M,N] = A[M,K] @ Bt[N,K]^T + bias.  One wave: 1 M-tile x 4 N-tiles
// (A fragment reused 4x). Block = 8 waves = 32 x 256 macro tile.
// OUTBF: store bf16; CTRANS: store bf16 transposed (Ct[n][m], ldc = M pitch).
template <bool OUTBF, bool CTRANS>
__global__ __launch_bounds__(256) void wmma_gemm_bf16(
    const __bf16* __restrict__ A, int lda,
    const __bf16* __restrict__ Bt, int ldbt,
    void* __restrict__ Cp, long ldc,
    int M, int N, int Kd, const float* __restrict__ bias) {
  const int wave = threadIdx.x >> 5;
  const int lane = threadIdx.x & 31;
  const int half = lane >> 4, lr = lane & 15;
  const int mbase = (blockIdx.y * 2 + (wave >> 2)) * 16;
  const int nbase = blockIdx.x * 256 + (wave & 3) * 64;
  if (mbase >= M || nbase >= N) return;   // wave-uniform

  const __bf16* arow = A + (long)(mbase + lr) * lda;
  v8f acc[4];
#pragma unroll
  for (int tt = 0; tt < 4; ++tt)
    acc[tt] = (v8f){0.f, 0.f, 0.f, 0.f, 0.f, 0.f, 0.f, 0.f};

  for (int k0 = 0; k0 < Kd; k0 += 32) {
    const v16bf af = ldfrag(arow + k0 + half * 8, 16);
#pragma unroll
    for (int tt = 0; tt < 4; ++tt) {
      const int col = nbase + tt * 16 + lr;
      const v16bf bf = ldfrag(Bt + (long)col * ldbt + k0 + half * 8, 16);
      acc[tt] = __builtin_amdgcn_wmma_f32_16x16x32_bf16(
          false, af, false, bf, (short)0, acc[tt], false, false);
    }
  }
#pragma unroll
  for (int tt = 0; tt < 4; ++tt) {
    const int col = nbase + tt * 16 + lr;
    const float bv = bias ? bias[col] : 0.f;
#pragma unroll
    for (int j = 0; j < 8; ++j) {
      const int row = mbase + half * 8 + j;
      const float v = acc[tt][j] + bv;
      if (CTRANS)     ((__bf16*)Cp)[(long)col * ldc + row] = f2bf(v);
      else if (OUTBF) ((__bf16*)Cp)[(long)row * ldc + col] = f2bf(v);
      else            ((float*)Cp)[(long)row * ldc + col] = v;
    }
  }
}

// --------------------- fused scores -> softmax -> ctx ----------------------
// Block = 8 waves, owns 16 query rows of one (b,h).
//  phase 1: each wave WMMAs a 16x256 score slice (16 tiles, Kd=128).
//  phase 2: block softmax (shfl_xor within 16-lane halves + LDS reduce);
//           attn -> HBM once (nontemporal), bf16 probs -> LDS strip.
//  phase 3: ctx[16,128] = strip @ V; wave w owns cols [16w,16w+16),
//           A fragments via conflict-free ds_load_b128 (row pitch 4112B).
__global__ __launch_bounds__(256) void mqa_fused(
    const __bf16* __restrict__ Qb,   // [B*S, H*Dh]
    const __bf16* __restrict__ Kb,   // [B*S, Dh]
    const __bf16* __restrict__ Vt,   // [Dh, B*S]
    float* __restrict__ attn,        // [B,H,S,S]
    __bf16* __restrict__ ctxb,       // [B*S, H*Dh]
    int S, int H, int MS, float scale) {
  constexpr int Dh = 128;
  const int z = blockIdx.z;
  const int b = z / H, h = z % H;
  const int m0 = blockIdx.x * 16;
  const int wave = threadIdx.x >> 5;
  const int lane = threadIdx.x & 31;
  const int half = lane >> 4, lr = lane & 15;
  const int D = H * Dh;

  __shared__ __bf16 strip[16][2048 + 8];   // +8 bf16 pad: 4-bank row shift
  __shared__ float red[16][8];
  __shared__ float rowstat[16];

  // Q fragments for the strip (Kd = 128 -> 4 slices of 32)
  const __bf16* qrow = Qb + (long)(b * S + m0 + lr) * D + h * Dh;
  v16bf qf[4];
#pragma unroll
  for (int ks = 0; ks < 4; ++ks)
    qf[ks] = ldfrag(qrow + ks * 32 + half * 8, 16);

  // ---- phase 1: scores ----
  v8f sacc[16];
  const int cbase = wave * 256;
#pragma unroll
  for (int t = 0; t < 16; ++t) {
    v8f a = (v8f){0.f, 0.f, 0.f, 0.f, 0.f, 0.f, 0.f, 0.f};
    const __bf16* krow = Kb + (long)(b * S + cbase + t * 16 + lr) * Dh;
#pragma unroll
    for (int ks = 0; ks < 4; ++ks) {
      const v16bf kf = ldfrag(krow + ks * 32 + half * 8, 16);
      a = __builtin_amdgcn_wmma_f32_16x16x32_bf16(
          false, qf[ks], false, kf, (short)0, a, false, false);
    }
#pragma unroll
    for (int j = 0; j < 8; ++j) a[j] *= scale;
    sacc[t] = a;
  }

  // ---- phase 2: softmax over all 2048 columns ----
  float pm[8];
#pragma unroll
  for (int j = 0; j < 8; ++j) pm[j] = -3.402823466e38f;
#pragma unroll
  for (int t = 0; t < 16; ++t)
#pragma unroll
    for (int j = 0; j < 8; ++j) pm[j] = fmaxf(pm[j], sacc[t][j]);
#pragma unroll
  for (int m = 1; m < 16; m <<= 1)
#pragma unroll
    for (int j = 0; j < 8; ++j) pm[j] = fmaxf(pm[j], __shfl_xor(pm[j], m, 32));
  if (lr == 0)
#pragma unroll
    for (int j = 0; j < 8; ++j) red[half * 8 + j][wave] = pm[j];
  __syncthreads();
  if (threadIdx.x < 16) {
    float mx = red[threadIdx.x][0];
    for (int w = 1; w < 8; ++w) mx = fmaxf(mx, red[threadIdx.x][w]);
    rowstat[threadIdx.x] = mx;
  }
  __syncthreads();
  float rm[8];
#pragma unroll
  for (int j = 0; j < 8; ++j) rm[j] = rowstat[half * 8 + j];

  float ps[8] = {0.f, 0.f, 0.f, 0.f, 0.f, 0.f, 0.f, 0.f};
#pragma unroll
  for (int t = 0; t < 16; ++t)
#pragma unroll
    for (int j = 0; j < 8; ++j) {
      const float e = __expf(sacc[t][j] - rm[j]);
      sacc[t][j] = e;
      ps[j] += e;
    }
#pragma unroll
  for (int m = 1; m < 16; m <<= 1)
#pragma unroll
    for (int j = 0; j < 8; ++j) ps[j] += __shfl_xor(ps[j], m, 32);
  __syncthreads();
  if (lr == 0)
#pragma unroll
    for (int j = 0; j < 8; ++j) red[half * 8 + j][wave] = ps[j];
  __syncthreads();
  if (threadIdx.x < 16) {
    float sm = 0.f;
    for (int w = 0; w < 8; ++w) sm += red[threadIdx.x][w];
    rowstat[threadIdx.x] = 1.f / sm;
  }
  __syncthreads();
  float inv[8];
#pragma unroll
  for (int j = 0; j < 8; ++j) inv[j] = rowstat[half * 8 + j];

  // normalize -> attn (streaming, single HBM pass) + bf16 LDS strip
  float* abase = attn + ((long)(b * H + h) * S + m0) * (long)S;
#pragma unroll
  for (int t = 0; t < 16; ++t) {
    const int c = cbase + t * 16 + lr;
#pragma unroll
    for (int j = 0; j < 8; ++j) {
      const float av = sacc[t][j] * inv[j];
      __builtin_nontemporal_store(av, abase + (long)(half * 8 + j) * S + c);
      strip[half * 8 + j][c] = f2bf(av);
    }
  }
  __syncthreads();

  // ---- phase 3: ctx = strip @ V ----
  const int n = wave * 16 + lr;
  const __bf16* vrow = Vt + (long)n * MS + (long)b * S;
  v8f cacc = (v8f){0.f, 0.f, 0.f, 0.f, 0.f, 0.f, 0.f, 0.f};
#pragma unroll 4
  for (int ks = 0; ks < 64; ++ks) {            // Kd = S = 2048
    const v16bf afr = ldfrag(&strip[lr][ks * 32 + half * 8], 16);
    const v16bf bfr = ldfrag(vrow + ks * 32 + half * 8, 16);
    cacc = __builtin_amdgcn_wmma_f32_16x16x32_bf16(
        false, afr, false, bfr, (short)0, cacc, false, false);
  }
#pragma unroll
  for (int j = 0; j < 8; ++j) {
    const int row = m0 + half * 8 + j;
    ctxb[(long)(b * S + row) * D + h * Dh + n] = f2bf(cacc[j]);
  }
}

// ------------------------------ driver -------------------------------------
extern "C" void kernel_launch(void* const* d_in, const int* in_sizes, int n_in,
                              void* d_out, int out_size, void* d_ws, size_t ws_size,
                              hipStream_t stream) {
  (void)in_sizes; (void)n_in; (void)out_size; (void)ws_size;

  const float* query = (const float*)d_in[0];
  const float* key_  = (const float*)d_in[1];
  const float* value = (const float*)d_in[2];
  const float* Wq = (const float*)d_in[3];
  const float* bq = (const float*)d_in[4];
  const float* Wk = (const float*)d_in[5];
  const float* bk = (const float*)d_in[6];
  const float* Wv = (const float*)d_in[7];
  const float* bv = (const float*)d_in[8];
  const float* Wo = (const float*)d_in[9];
  const float* bo = (const float*)d_in[10];

  const int  Bb = 2, S = 2048, D = 2048, H = 16, Dh = 128;
  const long MS = (long)Bb * S;                    // 4096

  float* out  = (float*)d_out;                     // [B*S, D]
  float* attn = out + MS * (long)D;                // [B,H,S,S]

  // workspace (bf16): ~105 MB total
  __bf16* w = (__bf16*)d_ws;
  auto take = [&](long nel) { __bf16* p = w; w += nel; return p; };
  __bf16* qcv = take(MS * (long)D);    // query bf16
  __bf16* kcv = take(MS * (long)D);    // key bf16
  __bf16* vcv = take(MS * (long)D);    // value bf16
  __bf16* Wqt = take((long)D * D);     // Wq^T bf16  [N=D][K=D]
  __bf16* Wkt = take((long)D * Dh);    // Wk^T bf16  [Dh][D]
  __bf16* Wvt = take((long)D * Dh);    // Wv^T bf16  [Dh][D]
  __bf16* Wot = take((long)D * D);     // Wo^T bf16  [D][D]
  __bf16* Qb  = take(MS * (long)D);    // Q bf16     [B*S][D]
  __bf16* Kbf = take(MS * (long)Dh);   // K bf16     [B*S][Dh]
  __bf16* Vtf = take(MS * (long)Dh);   // V^T bf16   [Dh][B*S]
  __bf16* Cb  = take(MS * (long)D);    // ctx bf16   [B*S][D]

  const dim3 blk(256);
  const long nact = MS * (long)D;                  // 8.4M elements

  // prep: one-time conversions / transposes
  cvt_bf16<<<dim3((unsigned)(nact / 1024)), blk, 0, stream>>>(query, qcv, nact);
  cvt_bf16<<<dim3((unsigned)(nact / 1024)), blk, 0, stream>>>(key_,  kcv, nact);
  cvt_bf16<<<dim3((unsigned)(nact / 1024)), blk, 0, stream>>>(value, vcv, nact);
  cvt_tr_bf16<<<dim3(D / 32,  D / 32), blk, 0, stream>>>(Wq, Wqt, D, D);
  cvt_tr_bf16<<<dim3(Dh / 32, D / 32), blk, 0, stream>>>(Wk, Wkt, D, Dh);
  cvt_tr_bf16<<<dim3(Dh / 32, D / 32), blk, 0, stream>>>(Wv, Wvt, D, Dh);
  cvt_tr_bf16<<<dim3(D / 32,  D / 32), blk, 0, stream>>>(Wo, Wot, D, D);

  // projections
  wmma_gemm_bf16<true, false><<<dim3(D / 256, MS / 32), blk, 0, stream>>>(
      qcv, D, Wqt, D, Qb, D, (int)MS, D, D, bq);
  wmma_gemm_bf16<true, false><<<dim3(1, MS / 32), blk, 0, stream>>>(
      kcv, D, Wkt, D, Kbf, Dh, (int)MS, Dh, D, bk);
  wmma_gemm_bf16<true, true><<<dim3(1, MS / 32), blk, 0, stream>>>(
      vcv, D, Wvt, D, Vtf, MS, (int)MS, Dh, D, bv);   // V stored transposed

  // fused attention: scores -> softmax -> attn(out) + ctx
  mqa_fused<<<dim3(S / 16, 1, Bb * H), blk, 0, stream>>>(
      Qb, Kbf, Vtf, attn, Cb, S, H, (int)MS, 0.08838834764831843f);

  // out = ctx @ Wo + bo
  wmma_gemm_bf16<false, false><<<dim3(D / 256, MS / 32), blk, 0, stream>>>(
      Cb, D, Wot, D, out, D, (int)MS, D, D, bo);
}
